// LinearMemoryAttention_16527034155640
// MI455X (gfx1250) — matrix operand order repaired and checked by reading, not compiled, not run on hardware
//
#include <hip/hip_runtime.h>
#include <hip/hip_bf16.h>

// ---------------------------------------------------------------------------
// Problem constants (B=2, S=4096, HID=4096, H=32, HKV=8, D=128, G=4)
// ---------------------------------------------------------------------------
#define BS_TOT   8192      // B*S
#define HID_DIM  4096
#define NKV      8
#define HD       128
#define GRP      4
#define KVW      1024      // NKV*HD
#define OUT_M_OFF 33554432 // BS_TOT*HID_DIM
#define OUT_Z_OFF 33685504 // + 8*128*128
#define MSLICES  8         // split-K slices for the M update

typedef __bf16 bf16_t;
typedef __attribute__((ext_vector_type(8)))  __bf16 bf16x8;
typedef __attribute__((ext_vector_type(16))) __bf16 bf16x16;
typedef __attribute__((ext_vector_type(8)))  float  f32x8;
typedef __attribute__((ext_vector_type(4)))  int    int4v;

#if __has_builtin(__builtin_amdgcn_global_load_async_to_lds_b128) && \
    __has_builtin(__builtin_amdgcn_s_wait_asynccnt)
#define USE_ASYNC_LDS 1
#else
#define USE_ASYNC_LDS 0
#endif

// 16-byte global -> LDS copy. Async path lowers to GLOBAL_LOAD_ASYNC_TO_LDS_B128
// (ASYNCcnt-tracked, no VGPR staging); fallback is load+ds_store.
__device__ __forceinline__ void copy16_to_lds(bf16_t* l, const bf16_t* g) {
#if USE_ASYNC_LDS
  __builtin_amdgcn_global_load_async_to_lds_b128(
      (__attribute__((address_space(1))) int4v*)g,
      (__attribute__((address_space(3))) int4v*)l, 0, 0);
#else
  *(bf16x8*)l = *(const bf16x8*)g;
#endif
}
__device__ __forceinline__ void async_wait_all() {
#if USE_ASYNC_LDS
  __builtin_amdgcn_s_wait_asynccnt(0);
#endif
}

__device__ __forceinline__ f32x8 wmma_bf16x32(bf16x16 a, bf16x16 b, f32x8 c) {
  return __builtin_amdgcn_wmma_f32_16x16x32_bf16(false, a, false, b, (short)0, c,
                                                 false, false);
}
__device__ __forceinline__ bf16x16 cat8(bf16x8 lo, bf16x8 hi) {
  return __builtin_shufflevector(lo, hi, 0,1,2,3,4,5,6,7,8,9,10,11,12,13,14,15);
}

// ---------------------------------------------------------------------------
// Generic bf16 GEMM:  C(MxN) = A(MxK, rm) * W(NxK, rm)^T [+ bias] [+ act]
// MODE 0: f32 store; MODE 1: bf16 store with sig=elu+1; MODE 2: bf16 store
// Block tile 128x128, BK=32, 256 threads = 8 waves, wave tile 64x32.
// Double-buffered LDS; tile k+1 is fetched (async where available) while
// tile k feeds the WMMAs.
// ---------------------------------------------------------------------------
template<int MODE>
__global__ __launch_bounds__(256)
void gemm_bf16_kernel(const bf16_t* __restrict__ A, const bf16_t* __restrict__ W,
                      const float* __restrict__ bias, void* __restrict__ Cout,
                      int M, int N, int K)
{
  __shared__ __align__(16) bf16_t As[2][128][40];
  __shared__ __align__(16) bf16_t Bs[2][128][40];
  const int t    = threadIdx.x;
  const int lane = t & 31, wv = t >> 5;
  const int wm0  = (wv >> 2) * 64, wn0 = (wv & 3) * 32;
  const int m0   = blockIdx.y * 128, n0 = blockIdx.x * 128;
  const int lm   = lane & 15, lh = lane >> 4;
  const int ka0  = lh ? 8 : 0;          // A-operand K split per ISA layout
  const int tq   = (t & 3) * 8, tr = t >> 2;

  const bf16_t* aR0 = A + (size_t)(m0 + tr     ) * K + tq;
  const bf16_t* aR1 = A + (size_t)(m0 + tr + 64) * K + tq;
  const bf16_t* wR0 = W + (size_t)(n0 + tr     ) * K + tq;
  const bf16_t* wR1 = W + (size_t)(n0 + tr + 64) * K + tq;

  auto stage = [&](int s, int k0) {
    copy16_to_lds(&As[s][tr     ][tq], aR0 + k0);
    copy16_to_lds(&As[s][tr + 64][tq], aR1 + k0);
    copy16_to_lds(&Bs[s][tr     ][tq], wR0 + k0);
    copy16_to_lds(&Bs[s][tr + 64][tq], wR1 + k0);
  };

  const f32x8 z8 = {0.f,0.f,0.f,0.f,0.f,0.f,0.f,0.f};
  f32x8 acc[4][2];
  #pragma unroll
  for (int i=0;i<4;i++) { acc[i][0] = z8; acc[i][1] = z8; }

  stage(0, 0);
  int s = 0;
  for (int k0 = 0; k0 < K; k0 += 32, s ^= 1) {
    async_wait_all();
    __syncthreads();
    if (k0 + 32 < K) stage(s ^ 1, k0 + 32);
    bf16x16 af[4], bfr[2];
    #pragma unroll
    for (int i=0;i<4;i++)
      af[i] = cat8(*(const bf16x8*)&As[s][wm0+i*16+lm][ka0],
                   *(const bf16x8*)&As[s][wm0+i*16+lm][ka0+16]);
    #pragma unroll
    for (int j=0;j<2;j++)
      bfr[j] = cat8(*(const bf16x8*)&Bs[s][wn0+j*16+lm][ka0],
                    *(const bf16x8*)&Bs[s][wn0+j*16+lm][ka0+16]);
    #pragma unroll
    for (int i=0;i<4;i++)
      #pragma unroll
      for (int j=0;j<2;j++)
        acc[i][j] = wmma_bf16x32(af[i], bfr[j], acc[i][j]);
  }

  #pragma unroll
  for (int i=0;i<4;i++) {
    const int gm = m0 + wm0 + i*16 + lh*8;
    #pragma unroll
    for (int j=0;j<2;j++) {
      const int gn = n0 + wn0 + j*16 + lm;
      const float bb = bias ? bias[gn] : 0.f;
      #pragma unroll
      for (int r=0;r<8;r++) {
        float val = acc[i][j][r] + bb;
        if (MODE == 1) val = (val >= 0.f) ? (val + 1.f) : __expf(val);
        const size_t off = (size_t)(gm + r) * N + gn;
        if (MODE == 0) ((float*)Cout)[off] = val;
        else           ((bf16_t*)Cout)[off] = (bf16_t)val;
      }
    }
  }
}

// ---------------------------------------------------------------------------
// Retrieval: per kv-head h  ret[m, h*128+e] = (sig_q_h[m,:] @ Mt_h[e,:]) / den
// den[m] = sig_q_h[m,:] . z_h + eps.  Mt is M pre-transposed to [h][e][d].
// ---------------------------------------------------------------------------
__global__ __launch_bounds__(256)
void retrieval_kernel(const bf16_t* __restrict__ sigq, const bf16_t* __restrict__ Mt,
                      const float* __restrict__ zv, bf16_t* __restrict__ ret)
{
  __shared__ __align__(16) bf16_t As[2][128][40];
  __shared__ __align__(16) bf16_t Bs[2][128][40];
  __shared__ float den[128];
  const int h  = blockIdx.y;
  const int m0 = blockIdx.x * 128;
  const int t  = threadIdx.x;
  const int lane = t & 31, wv = t >> 5;
  const int wm0 = (wv >> 2) * 64, wn0 = (wv & 3) * 32;
  const int lm  = lane & 15, lh = lane >> 4;
  const int ka0 = lh ? 8 : 0;
  const int tq  = (t & 3) * 8, tr = t >> 2;

  const bf16_t* aR0 = sigq + (size_t)(m0 + tr     ) * KVW + h * HD + tq;
  const bf16_t* aR1 = sigq + (size_t)(m0 + tr + 64) * KVW + h * HD + tq;
  const bf16_t* wR0 = Mt + (size_t)h*HD*HD + (size_t)(tr     ) * HD + tq;
  const bf16_t* wR1 = Mt + (size_t)h*HD*HD + (size_t)(tr + 64) * HD + tq;

  auto stage = [&](int s, int k0) {
    copy16_to_lds(&As[s][tr     ][tq], aR0 + k0);
    copy16_to_lds(&As[s][tr + 64][tq], aR1 + k0);
    copy16_to_lds(&Bs[s][tr     ][tq], wR0 + k0);
    copy16_to_lds(&Bs[s][tr + 64][tq], wR1 + k0);
  };

  if (t < 128) {
    const bf16_t* row = sigq + (size_t)(m0 + t) * KVW + h * HD;
    float ss = 0.f;
    for (int d = 0; d < HD; ++d) ss += (float)row[d] * zv[h*HD + d];
    den[t] = ss + 1e-6f;
  }

  const f32x8 z8 = {0.f,0.f,0.f,0.f,0.f,0.f,0.f,0.f};
  f32x8 acc[4][2];
  #pragma unroll
  for (int i=0;i<4;i++) { acc[i][0] = z8; acc[i][1] = z8; }

  stage(0, 0);
  int s = 0;
  for (int k0 = 0; k0 < HD; k0 += 32, s ^= 1) {
    async_wait_all();
    __syncthreads();
    if (k0 + 32 < HD) stage(s ^ 1, k0 + 32);
    bf16x16 af[4], bfr[2];
    #pragma unroll
    for (int i=0;i<4;i++)
      af[i] = cat8(*(const bf16x8*)&As[s][wm0+i*16+lm][ka0],
                   *(const bf16x8*)&As[s][wm0+i*16+lm][ka0+16]);
    #pragma unroll
    for (int j=0;j<2;j++)
      bfr[j] = cat8(*(const bf16x8*)&Bs[s][wn0+j*16+lm][ka0],
                    *(const bf16x8*)&Bs[s][wn0+j*16+lm][ka0+16]);
    #pragma unroll
    for (int i=0;i<4;i++)
      #pragma unroll
      for (int j=0;j<2;j++)
        acc[i][j] = wmma_bf16x32(af[i], bfr[j], acc[i][j]);
  }

  #pragma unroll
  for (int i=0;i<4;i++) {
    const int rloc = wm0 + i*16 + lh*8;
    #pragma unroll
    for (int j=0;j<2;j++) {
      const int e = wn0 + j*16 + lm;
      #pragma unroll
      for (int r=0;r<8;r++) {
        float val = acc[i][j][r] / den[rloc + r];
        ret[(size_t)(m0 + rloc + r)*KVW + h*HD + e] = (bf16_t)val;
      }
    }
  }
}

// ---------------------------------------------------------------------------
// M-update partials: part[slice,h,d,e] = sum_{m in slice} k[m,d] * v[m,e]
// Grid (NKV, MSLICES). 32x128 chunks transposed into LDS so both WMMA
// operands read with the standard contiguous fragment pattern.
// ---------------------------------------------------------------------------
__global__ __launch_bounds__(256)
void mupdate_kernel(const bf16_t* __restrict__ sigk, const bf16_t* __restrict__ vb,
                    float* __restrict__ part)
{
  __shared__ __align__(16) bf16_t Ksm[128][40];   // [d][m_local]
  __shared__ __align__(16) bf16_t Vsm[128][40];   // [e][m_local]
  const int h  = blockIdx.x;
  const int sl = blockIdx.y;
  const int t = threadIdx.x;
  const int lane = t & 31, wv = t >> 5;
  const int wm0 = (wv >> 2) * 64, wn0 = (wv & 3) * 32;
  const int lm  = lane & 15, lh = lane >> 4;
  const int ka0 = lh ? 8 : 0;
  const int ml  = t & 31;            // m within chunk
  const int dg  = (t >> 5) * 16;     // 16-wide d/e group

  const f32x8 z8 = {0.f,0.f,0.f,0.f,0.f,0.f,0.f,0.f};
  f32x8 acc[4][2];
  #pragma unroll
  for (int i=0;i<4;i++) { acc[i][0] = z8; acc[i][1] = z8; }

  const int b_lo = sl * (BS_TOT / MSLICES);
  const int b_hi = b_lo + (BS_TOT / MSLICES);
  for (int b0 = b_lo; b0 < b_hi; b0 += 32) {
    __syncthreads();
    const size_t base = (size_t)(b0 + ml) * KVW + h * HD + dg;
    bf16x8 k0v = *(const bf16x8*)&sigk[base];
    bf16x8 k1v = *(const bf16x8*)&sigk[base + 8];
    bf16x8 v0v = *(const bf16x8*)&vb[base];
    bf16x8 v1v = *(const bf16x8*)&vb[base + 8];
    #pragma unroll
    for (int i=0;i<8;i++) {
      Ksm[dg + i    ][ml] = k0v[i];
      Ksm[dg + 8 + i][ml] = k1v[i];
      Vsm[dg + i    ][ml] = v0v[i];
      Vsm[dg + 8 + i][ml] = v1v[i];
    }
    __syncthreads();
    bf16x16 af[4], bfr[2];
    #pragma unroll
    for (int i=0;i<4;i++)
      af[i] = cat8(*(const bf16x8*)&Ksm[wm0+i*16+lm][ka0],
                   *(const bf16x8*)&Ksm[wm0+i*16+lm][ka0+16]);
    #pragma unroll
    for (int j=0;j<2;j++)
      bfr[j] = cat8(*(const bf16x8*)&Vsm[wn0+j*16+lm][ka0],
                    *(const bf16x8*)&Vsm[wn0+j*16+lm][ka0+16]);
    #pragma unroll
    for (int i=0;i<4;i++)
      #pragma unroll
      for (int j=0;j<2;j++)
        acc[i][j] = wmma_bf16x32(af[i], bfr[j], acc[i][j]);
  }

  float* dst = part + (size_t)sl * (NKV*HD*HD) + (size_t)h * HD*HD;
  #pragma unroll
  for (int i=0;i<4;i++) {
    const int d = wm0 + i*16 + lh*8;
    #pragma unroll
    for (int j=0;j<2;j++) {
      const int e = wn0 + j*16 + lm;
      #pragma unroll
      for (int r=0;r<8;r++)
        dst[(size_t)(d + r)*HD + e] = acc[i][j][r];
    }
  }
}

__global__ void mfinal_kernel(const float* __restrict__ part,
                              const float* __restrict__ Mold,
                              float* __restrict__ Mout) {
  const int i = blockIdx.x * blockDim.x + threadIdx.x;
  if (i >= NKV*HD*HD) return;
  float sacc = Mold[i];
  #pragma unroll
  for (int sl = 0; sl < MSLICES; ++sl) sacc += part[sl * (NKV*HD*HD) + i];
  Mout[i] = sacc;
}

// ---------------------------------------------------------------------------
// z_new two-stage deterministic reduction
// ---------------------------------------------------------------------------
__global__ __launch_bounds__(256)
void znew_partial(const bf16_t* __restrict__ sigk, float* __restrict__ part) {
  const int b = blockIdx.x;      // 64 blocks x 128 rows
  const int t = threadIdx.x;
  float a0=0.f, a1=0.f, a2=0.f, a3=0.f;
  const bf16_t* base = sigk + (size_t)b * 128 * KVW;
  for (int r = 0; r < 128; ++r) {
    const bf16_t* row = base + (size_t)r * KVW;
    a0 += (float)row[t      ];
    a1 += (float)row[t + 256];
    a2 += (float)row[t + 512];
    a3 += (float)row[t + 768];
  }
  part[b*KVW + t      ] = a0;
  part[b*KVW + t + 256] = a1;
  part[b*KVW + t + 512] = a2;
  part[b*KVW + t + 768] = a3;
}

__global__ void znew_final(const float* __restrict__ part,
                           const float* __restrict__ zv, float* __restrict__ zout) {
  const int t = blockIdx.x * blockDim.x + threadIdx.x;
  if (t >= KVW) return;
  float sacc = zv[t];
  for (int b = 0; b < 64; ++b) sacc += part[b*KVW + t];
  zout[t] = sacc;
}

// ---------------------------------------------------------------------------
// Prep kernels (f32 -> bf16 packing, row gather, Wo fold, M transpose)
// ---------------------------------------------------------------------------
__global__ void cvt_f32_bf16(const float* __restrict__ src, bf16_t* __restrict__ dst,
                             long long n) {
  long long i = (long long)blockIdx.x * blockDim.x + threadIdx.x;
  const long long stride = (long long)gridDim.x * blockDim.x;
  for (; i < n; i += stride) dst[i] = (bf16_t)src[i];
}

__global__ void gather_wq(const float* __restrict__ Wq, bf16_t* __restrict__ dst) {
  const long long n = (long long)KVW * HID_DIM;
  long long i = (long long)blockIdx.x * blockDim.x + threadIdx.x;
  const long long stride = (long long)gridDim.x * blockDim.x;
  for (; i < n; i += stride) {
    const long long r = i >> 12, c = i & 4095;
    const long long hk = r >> 7, d = r & 127;
    dst[i] = (bf16_t)Wq[(hk * (GRP*HD) + d) * HID_DIM + c];
  }
}

__global__ void gather_bq(const float* __restrict__ bq, float* __restrict__ dst) {
  const int t = blockIdx.x * blockDim.x + threadIdx.x;
  if (t < KVW) {
    const int hk = t >> 7, d = t & 127;
    dst[t] = bq[hk * (GRP*HD) + d];
  }
}

__global__ void fold_wo(const float* __restrict__ Wo, bf16_t* __restrict__ dst) {
  const long long n = (long long)HID_DIM * KVW;
  long long i = (long long)blockIdx.x * blockDim.x + threadIdx.x;
  const long long stride = (long long)gridDim.x * blockDim.x;
  for (; i < n; i += stride) {
    const long long o = i >> 10, c = i & 1023;
    const long long hk = c >> 7, d = c & 127;
    float sacc = 0.f;
    #pragma unroll
    for (int g = 0; g < GRP; ++g)
      sacc += Wo[o * HID_DIM + (hk*GRP + g) * HD + d];
    dst[i] = (bf16_t)sacc;
  }
}

__global__ void transpose_m(const float* __restrict__ Mm, bf16_t* __restrict__ Mt) {
  const int n = NKV * HD * HD;
  int i = blockIdx.x * blockDim.x + threadIdx.x;
  const int stride = gridDim.x * blockDim.x;
  for (; i < n; i += stride) {
    const int h = i >> 14, rem = i & 16383;
    const int e = rem >> 7, d = rem & 127;
    Mt[i] = (bf16_t)Mm[(h << 14) + d * HD + e];
  }
}

// ---------------------------------------------------------------------------
// Launch
// ---------------------------------------------------------------------------
extern "C" void kernel_launch(void* const* d_in, const int* in_sizes, int n_in,
                              void* d_out, int out_size, void* d_ws, size_t ws_size,
                              hipStream_t stream) {
  (void)in_sizes; (void)n_in; (void)out_size; (void)ws_size;
  const float* hs = (const float*)d_in[0];
  const float* Wq = (const float*)d_in[1];
  const float* bq = (const float*)d_in[2];
  const float* Wk = (const float*)d_in[3];
  const float* bk = (const float*)d_in[4];
  const float* Wv = (const float*)d_in[5];
  const float* bv = (const float*)d_in[6];
  const float* Wo = (const float*)d_in[7];
  const float* Mm = (const float*)d_in[8];
  const float* zv = (const float*)d_in[9];
  float* out = (float*)d_out;

  char* ws = (char*)d_ws;
  bf16_t* h_bf  = (bf16_t*)(ws);                    // 8192x4096   (67108864 B)
  bf16_t* wq_bf = (bf16_t*)(ws + 67108864);         // 1024x4096   ( 8388608 B)
  bf16_t* wk_bf = (bf16_t*)(ws + 75497472);         // 1024x4096
  bf16_t* wv_bf = (bf16_t*)(ws + 83886080);         // 1024x4096
  bf16_t* wo_bf = (bf16_t*)(ws + 92274688);         // 4096x1024 (folded Wo)
  bf16_t* sigq  = (bf16_t*)(ws + 100663296);        // 8192x1024  (16777216 B)
  bf16_t* sigk  = (bf16_t*)(ws + 117440512);        // 8192x1024
  bf16_t* vbuf  = (bf16_t*)(ws + 134217728);        // 8192x1024
  bf16_t* ret   = (bf16_t*)(ws + 150994944);        // 8192x1024
  bf16_t* mt_bf = (bf16_t*)(ws + 167772160);        // 8x128x128   (262144 B)
  float*  bqsel = (float*)(ws + 168034304);         // 1024 f32    (4096 B)
  float*  mpart = (float*)(ws + 168038400);         // 8x8x128x128 (4194304 B)
  float*  zpart = (float*)(ws + 172232704);         // 64x1024 f32 (262144 B)

  // --- prep / precision packing ---
  cvt_f32_bf16<<<2048, 256, 0, stream>>>(hs, h_bf, (long long)BS_TOT * HID_DIM);
  gather_wq   <<<1024, 256, 0, stream>>>(Wq, wq_bf);
  cvt_f32_bf16<<<1024, 256, 0, stream>>>(Wk, wk_bf, (long long)KVW * HID_DIM);
  cvt_f32_bf16<<<1024, 256, 0, stream>>>(Wv, wv_bf, (long long)KVW * HID_DIM);
  fold_wo     <<<1024, 256, 0, stream>>>(Wo, wo_bf);
  gather_bq   <<<4, 256, 0, stream>>>(bq, bqsel);
  transpose_m <<<128, 256, 0, stream>>>(Mm, mt_bf);

  // --- projections (WMMA, async double-buffered LDS) ---
  dim3 gQKV(KVW / 128, BS_TOT / 128);               // (8, 64)
  gemm_bf16_kernel<1><<<gQKV, 256, 0, stream>>>(h_bf, wq_bf, bqsel, (void*)sigq,
                                                BS_TOT, KVW, HID_DIM);
  gemm_bf16_kernel<1><<<gQKV, 256, 0, stream>>>(h_bf, wk_bf, bk, (void*)sigk,
                                                BS_TOT, KVW, HID_DIM);
  gemm_bf16_kernel<2><<<gQKV, 256, 0, stream>>>(h_bf, wv_bf, bv, (void*)vbuf,
                                                BS_TOT, KVW, HID_DIM);

  // --- retrieval (WMMA, per kv head) ---
  dim3 gRet(BS_TOT / 128, NKV);                     // (64, 8)
  retrieval_kernel<<<gRet, 256, 0, stream>>>(sigq, mt_bf, zv, ret);

  // --- output projection with folded Wo (WMMA) ---
  dim3 gOut(HID_DIM / 128, BS_TOT / 128);           // (32, 64)
  gemm_bf16_kernel<0><<<gOut, 256, 0, stream>>>(ret, wo_bf, nullptr, (void*)out,
                                                BS_TOT, HID_DIM, KVW);

  // --- state updates (deterministic split-K + tree reduce) ---
  dim3 gMu(NKV, MSLICES);                           // (8, 8)
  mupdate_kernel<<<gMu, 256, 0, stream>>>(sigk, vbuf, mpart);
  mfinal_kernel<<<(NKV*HD*HD + 255)/256, 256, 0, stream>>>(mpart, Mm, out + OUT_M_OFF);
  znew_partial<<<64, 256, 0, stream>>>(sigk, zpart);
  znew_final<<<4, 256, 0, stream>>>(zpart, zv, out + OUT_Z_OFF);
}
